// ScenePerceiver_85710367359122
// MI455X (gfx1250) — matrix operand order repaired and checked by reading, not compiled, not run on hardware
//
#include <hip/hip_runtime.h>
#include <cstdint>
#include <cstddef>

// ---------------- model constants ----------------
#define NB      8
#define NKV     16384
#define NDIM    240
#define NDIMP   256        // K-dim padded for 16x16x32 bf16 WMMA
#define NHEADS  8
#define HDIM    30
#define NTOKS   16
#define NFFN    1024
#define NLAY    4

// attention tiling: 4 waves/WG, each wave does 8 tiles of 16 kv rows = 128 rows
#define WAVES_PER_WG   4
#define TILES_PER_WAVE 8
#define ROWS_PER_WAVE  (TILES_PER_WAVE*16)          // 128
#define ROWS_PER_WG    (WAVES_PER_WG*ROWS_PER_WAVE) // 512
#define NCHUNK         (NKV/ROWS_PER_WAVE)          // 128 wave-chunks per batch
#define WG_PER_B       (NKV/ROWS_PER_WG)            // 32

// per-wave LDS carve (bytes)
#define OFF_ASTAGE 0            // bf16 [16][256]  (also reused as Krot)
#define OFF_KF     8192         // f32  [16][240]
#define OFF_VF     23552        // f32  [16][240]
#define OFF_PBUF   38912        // f32  [8][16][16]
#define OFF_ALBUF  47104        // f32  [8][16]
#define OFF_MROW   47616        // f32  [8][16]
#define OFF_LROW   48128        // f32  [8][16]
#define OFF_FACC   48640        // f32  [16][240]
#define WAVE_LDS   64000
#define ATTN_LDS   (WAVE_LDS*WAVES_PER_WG)          // 256000 B (<320KB WG limit)

#define EPI_LDS    131072
#define Q_LDS      16384

typedef __bf16 bf16_t;
typedef __bf16 v16bf __attribute__((ext_vector_type(16)));
typedef float  v8f   __attribute__((ext_vector_type(8)));

union Frag { v16bf v; uint4 q[2]; };

// inv_freq = 10000^{-j/5}, j = 0..4
__device__ const float c_invf[5] = {1.0f, 0.15848932f, 0.025118865f,
                                    0.0039810718f, 0.00063095737f};

__device__ inline v8f wmma_bf16(v16bf a, v16bf b, v8f c) {
  // D = A(16x32 bf16) x B(32x16 bf16) + C(16x16 f32)
  return __builtin_amdgcn_wmma_f32_16x16x32_bf16(false, a, false, b,
                                                 (short)0, c, false, false);
}

// A-fragment (contiguous 16-bit A layout): lane m = lane%16 holds row m.
// g = lane/16. elements 0..15 -> K = kb+16g..+15
__device__ inline v16bf loadA_lds(const bf16_t* base, int m, int kb, int g) {
  Frag f;
  const uint4* p0 = (const uint4*)(base + m*NDIMP + kb + 16*g);
  const uint4* p1 = (const uint4*)(base + m*NDIMP + kb + 16*g + 8);
  f.q[0] = *p0; f.q[1] = *p1;
  return f.v;
}

// B-fragment (contiguous 16-bit B layout, same k order as A): lane holds column n = lane%16, 16
// consecutive K values starting at kb + 16*(lane/16).
__device__ inline v16bf loadB_glb(const bf16_t* rowp, int kb, int g) {
  Frag f;
  const uint4* p = (const uint4*)(rowp + kb + 16*g);
  f.q[0] = p[0]; f.q[1] = p[1];
  return f.v;
}

__device__ inline v8f v8f_zero() {
  v8f z;
#pragma unroll
  for (int i = 0; i < 8; ++i) z[i] = 0.f;
  return z;
}

// ---------------------------------------------------------------------------
// x[b][t][d] = query_tokens[t][d]
__global__ __launch_bounds__(256) void init_x_kernel(
    const float* __restrict__ qt, float* __restrict__ x) {
  int b = blockIdx.x;
  for (int i = threadIdx.x; i < NTOKS*NDIM; i += 256)
    x[(size_t)b*NTOKS*NDIM + i] = qt[i];
}

// Convert a [240][240] f32 weight to bf16 [240][256] (K padded with zeros)
__global__ __launch_bounds__(256) void conv_w_kernel(
    const float* __restrict__ W, bf16_t* __restrict__ Wb) {
  int row = blockIdx.x;
  int c   = threadIdx.x;               // 0..255
  float v = (c < NDIM) ? W[(size_t)row*NDIM + c] : 0.f;
  Wb[(size_t)row*NDIMP + c] = (bf16_t)v;
}

// ---------------------------------------------------------------------------
// q = x @ Wq^T + bq, rotary with head_trans coords, packed into WMMA B-operand
// layout: qB[b][h][lane][e] = q_rot[tok = lane%16][dim = e + 16*(lane/16)]
__global__ __launch_bounds__(256) void q_kernel(
    const float* __restrict__ x, const float* __restrict__ Wq,
    const float* __restrict__ bq, const float* __restrict__ head_trans,
    bf16_t* __restrict__ qB) {
  extern __shared__ char smem[];
  float* qf = (float*)smem;            // [16][240]
  int b = blockIdx.x, t = threadIdx.x;

  for (int i = t; i < NTOKS*NDIM; i += 256) {
    int tok = i / NDIM, o = i % NDIM;
    float s = bq[o];
    const float* xr = x + ((size_t)b*NTOKS + tok)*NDIM;
    const float* wr = Wq + (size_t)o*NDIM;
    for (int k = 0; k < NDIM; ++k) s += xr[k]*wr[k];
    qf[i] = s;
  }
  __syncthreads();
  // 3D rotary, in place (each thread owns whole pairs)
  for (int i = t; i < NTOKS*120; i += 256) {
    int tok = i / 120, pr = i % 120;
    int h = pr / 15, ip = pr % 15, ax = ip / 5, j = ip % 5;
    int c1 = tok*NDIM + h*HDIM + ax*10 + 2*j;
    float x1 = qf[c1], x2 = qf[c1+1];
    float th = head_trans[b*3 + ax] * c_invf[j];
    float cs = __cosf(th), sn = __sinf(th);
    qf[c1]   = x1*cs - x2*sn;
    qf[c1+1] = x1*sn + x2*cs;
  }
  __syncthreads();
  for (int i = t; i < NHEADS*32*16; i += 256) {
    int h = i >> 9, rem = i & 511, l = rem >> 4, e = rem & 15;
    int tok = l & 15, gg = l >> 4;
    int kd  = e + 16*gg;
    float v = (kd < HDIM) ? qf[tok*NDIM + h*HDIM + kd] : 0.f;
    qB[(((size_t)b*NHEADS + h)*32 + l)*16 + e] = (bf16_t)v;
  }
}

// ---------------------------------------------------------------------------
// Fused flash cross-attention chunk kernel.
// grid: NB*WG_PER_B blocks of 128 threads (4 wave32). Each wave streams 128
// kv rows (8 tiles of 16): bf16 stage -> WMMA K/V projections -> rotary(K) ->
// per-head WMMA score tile vs packed q -> online softmax -> PV accumulate.
__global__ __launch_bounds__(128) void attn_kernel(
    const float* __restrict__ kv_feats, const float* __restrict__ kv_coords,
    const unsigned char* __restrict__ kv_mask,
    const bf16_t* __restrict__ WkB, const bf16_t* __restrict__ WvB,
    const float* __restrict__ bk, const float* __restrict__ bv,
    const bf16_t* __restrict__ qB,
    float* __restrict__ pm, float* __restrict__ pl, float* __restrict__ pa) {
  extern __shared__ char smem[];
  const int b    = blockIdx.x / WG_PER_B;
  const int cg   = blockIdx.x % WG_PER_B;
  const int w    = threadIdx.x >> 5;
  const int lane = threadIdx.x & 31;
  const int g    = lane >> 4;
  const int ln   = lane & 15;

  char* wb = smem + w*WAVE_LDS;
  bf16_t* astage = (bf16_t*)(wb + OFF_ASTAGE);   // also Krot after V GEMM
  float*  Kf     = (float*)(wb + OFF_KF);
  float*  Vf     = (float*)(wb + OFF_VF);
  float*  pbuf   = (float*)(wb + OFF_PBUF);
  float*  albuf  = (float*)(wb + OFF_ALBUF);
  float*  mrow   = (float*)(wb + OFF_MROW);
  float*  lrow   = (float*)(wb + OFF_LROW);
  float*  facc   = (float*)(wb + OFF_FACC);

  for (int i = lane; i < NTOKS*NDIM; i += 32) facc[i] = 0.f;
  for (int i = lane; i < 128; i += 32) { mrow[i] = -1e30f; lrow[i] = 0.f; }

  const float scale = 0.18257418583505537f;  // 1/sqrt(30)

  for (int t = 0; t < TILES_PER_WAVE; ++t) {
    const int row0 = cg*ROWS_PER_WG + w*ROWS_PER_WAVE + t*16;
    __syncthreads();
    // ---- stage kv tile as bf16 [16][256] (+ prefetch next tile) ----
    if (t + 1 < TILES_PER_WAVE) {
      const float* nxt =
          kv_feats + ((size_t)b*NKV + row0 + 16 + (lane >> 1))*NDIM + (lane & 1)*128;
      __builtin_prefetch(nxt, 0, 1);
    }
    for (int i = lane; i < 16*NDIM; i += 32) {
      int r = i / NDIM, c = i % NDIM;
      astage[r*NDIMP + c] =
          (bf16_t)kv_feats[((size_t)b*NKV + row0 + r)*NDIM + c];
    }
    for (int i = lane; i < 16*16; i += 32) {
      int r = i >> 4, c = NDIM + (i & 15);
      astage[r*NDIMP + c] = (bf16_t)0.f;
    }
    __syncthreads();

    // ---- K projection: 8 k-steps x 15 n-tiles of WMMA ----
    v8f acc[15];
#pragma unroll
    for (int n = 0; n < 15; ++n) acc[n] = v8f_zero();
    for (int kk = 0; kk < 8; ++kk) {
      v16bf fa = loadA_lds(astage, ln, kk*32, g);
#pragma unroll
      for (int n = 0; n < 15; ++n) {
        v16bf fb = loadB_glb(WkB + (size_t)(n*16 + ln)*NDIMP, kk*32, g);
        acc[n] = wmma_bf16(fa, fb, acc[n]);
      }
    }
#pragma unroll
    for (int n = 0; n < 15; ++n) {
      float bias = bk[n*16 + ln];
#pragma unroll
      for (int r = 0; r < 8; ++r)
        Kf[(r + 8*g)*NDIM + n*16 + ln] = acc[n][r] + bias;
    }

    // ---- V projection ----
#pragma unroll
    for (int n = 0; n < 15; ++n) acc[n] = v8f_zero();
    for (int kk = 0; kk < 8; ++kk) {
      v16bf fa = loadA_lds(astage, ln, kk*32, g);
#pragma unroll
      for (int n = 0; n < 15; ++n) {
        v16bf fb = loadB_glb(WvB + (size_t)(n*16 + ln)*NDIMP, kk*32, g);
        acc[n] = wmma_bf16(fa, fb, acc[n]);
      }
    }
#pragma unroll
    for (int n = 0; n < 15; ++n) {
      float bias = bv[n*16 + ln];
#pragma unroll
      for (int r = 0; r < 8; ++r)
        Vf[(r + 8*g)*NDIM + n*16 + ln] = acc[n][r] + bias;
    }
    __syncthreads();

    // ---- rotary on K, write bf16 Krot over astage: head h at cols h*32 ----
    for (int i = lane; i < 16*120; i += 32) {
      int r = i / 120, pr = i % 120;
      int h = pr / 15, ip = pr % 15, ax = ip / 5, j = ip % 5;
      int c1 = h*HDIM + ax*10 + 2*j;
      float x1 = Kf[r*NDIM + c1], x2 = Kf[r*NDIM + c1 + 1];
      float coord = kv_coords[((size_t)b*NKV + row0 + r)*3 + ax];
      float th = coord * c_invf[j];
      float cs = __cosf(th), sn = __sinf(th);
      int dst = r*NDIMP + h*32 + ax*10 + 2*j;
      astage[dst]     = (bf16_t)(x1*cs - x2*sn);
      astage[dst + 1] = (bf16_t)(x1*sn + x2*cs);
    }
    for (int i = lane; i < 16*8; i += 32) {      // zero pad dims 30,31
      int r = i >> 3, h = i & 7;
      astage[r*NDIMP + h*32 + 30] = (bf16_t)0.f;
      astage[r*NDIMP + h*32 + 31] = (bf16_t)0.f;
    }
    __syncthreads();

    bool mk[8];
#pragma unroll
    for (int r = 0; r < 8; ++r)
      mk[r] = kv_mask[(size_t)b*NKV + row0 + r + 8*g] != 0;

    // ---- scores per head: one WMMA each, online softmax update ----
#pragma unroll
    for (int h = 0; h < NHEADS; ++h) {
      v16bf fa = loadA_lds(astage, ln, h*32, g);
      Frag fq;
      const uint4* qp =
          (const uint4*)(qB + (((size_t)b*NHEADS + h)*32 + lane)*16);
      fq.q[0] = qp[0]; fq.q[1] = qp[1];
      v8f s = wmma_bf16(fa, fq.v, v8f_zero());   // D[kv=r+8g][tok=ln]

      float sv[8];
      float tm = -1e30f;
#pragma unroll
      for (int r = 0; r < 8; ++r) {
        float xx = mk[r] ? -1e30f : s[r]*scale;
        sv[r] = xx;
        tm = fmaxf(tm, xx);
      }
      tm = fmaxf(tm, __shfl_xor(tm, 16));
      float mold = mrow[h*16 + ln];
      float nm   = fmaxf(mold, tm);
      float ts = 0.f;
#pragma unroll
      for (int r = 0; r < 8; ++r) {
        float p = __expf(sv[r] - nm);
        pbuf[h*256 + (r + 8*g)*16 + ln] = p;
        ts += p;
      }
      ts += __shfl_xor(ts, 16);
      if (g == 0) {
        float al = __expf(mold - nm);
        albuf[h*16 + ln] = al;
        mrow[h*16 + ln]  = nm;
        lrow[h*16 + ln]  = lrow[h*16 + ln]*al + ts;
      }
    }
    __syncthreads();

    // ---- PV accumulate: lane owns q = ln, column half g ----
    {
      int q = ln;
      for (int h = 0; h < NHEADS; ++h) {
        float p[16];
#pragma unroll
        for (int kv = 0; kv < 16; ++kv) p[kv] = pbuf[h*256 + kv*16 + q];
        float al = albuf[h*16 + q];
        for (int cc = 0; cc < 15; ++cc) {
          int c = h*HDIM + g*15 + cc;
          float sacc = 0.f;
#pragma unroll
          for (int kv = 0; kv < 16; ++kv) sacc += p[kv]*Vf[kv*NDIM + c];
          facc[q*NDIM + c] = facc[q*NDIM + c]*al + sacc;
        }
      }
    }
  }
  __syncthreads();

  // ---- write flash partials ----
  const int cw = cg*WAVES_PER_WG + w;            // 0..127 per batch
  for (int i = lane; i < 128; i += 32) {
    pm[((size_t)b*NCHUNK + cw)*128 + i] = mrow[i];
    pl[((size_t)b*NCHUNK + cw)*128 + i] = lrow[i];
  }
  for (int i = lane; i < NTOKS*NDIM; i += 32)
    pa[((size_t)b*NCHUNK + cw)*NTOKS*NDIM + i] = facc[i];
}

// ---------------------------------------------------------------------------
__device__ void lnorm_block(const float* in, float* out,
                            const float* gg, const float* bb,
                            float* st, int t) {
  if (t < NTOKS) {
    float m = 0.f;
    for (int k = 0; k < NDIM; ++k) m += in[t*NDIM + k];
    m /= (float)NDIM;
    float v = 0.f;
    for (int k = 0; k < NDIM; ++k) { float d = in[t*NDIM + k] - m; v += d*d; }
    v /= (float)NDIM;
    st[t]         = m;
    st[NTOKS + t] = rsqrtf(v + 1e-5f);
  }
  __syncthreads();
  for (int i = t; i < NTOKS*NDIM; i += 256) {
    int q = i / NDIM, c = i % NDIM;
    out[i] = (in[i] - st[q]) * st[NTOKS + q] * gg[c] + bb[c];
  }
}

// merge flash partials + Wo proj + LN + FFN(GELU exact) + LN, updates x
__global__ __launch_bounds__(256) void epilogue_kernel(
    const float* __restrict__ pm, const float* __restrict__ pl,
    const float* __restrict__ pa,
    const float* __restrict__ Wo,  const float* __restrict__ bo,
    const float* __restrict__ W1,  const float* __restrict__ b1,
    const float* __restrict__ W2,  const float* __restrict__ b2,
    const float* __restrict__ ln1g, const float* __restrict__ ln1b,
    const float* __restrict__ ln2g, const float* __restrict__ ln2b,
    float* __restrict__ x) {
  extern __shared__ char smem[];
  float* big  = (float*)smem;                 // 16384 f32: sf[128][128] / g1[16][1024]
  float* abuf = big + 16384;                  // [16][240]
  float* xbuf = abuf + NTOKS*NDIM;
  float* hbuf = xbuf + NTOKS*NDIM;
  float* tbuf = hbuf + NTOKS*NDIM;
  float* Mh   = tbuf + NTOKS*NDIM;            // [128]
  float* Lh   = Mh + 128;                     // [128]
  float* st   = Lh + 128;                     // [32]
  int b = blockIdx.x, t = threadIdx.x;

  for (int i = t; i < NTOKS*NDIM; i += 256)
    xbuf[i] = x[(size_t)b*NTOKS*NDIM + i];

  if (t < 128) {                              // global max / normalizer per (h,q)
    float m = -1e30f;
    for (int cw = 0; cw < NCHUNK; ++cw)
      m = fmaxf(m, pm[((size_t)b*NCHUNK + cw)*128 + t]);
    float l = 0.f;
    for (int cw = 0; cw < NCHUNK; ++cw)
      l += pl[((size_t)b*NCHUNK + cw)*128 + t] *
           __expf(pm[((size_t)b*NCHUNK + cw)*128 + t] - m);
    Mh[t] = m; Lh[t] = l;
  }
  __syncthreads();
  for (int i = t; i < NCHUNK*128; i += 256) { // rescale factors
    int cw = i >> 7, hq = i & 127;
    big[i] = __expf(pm[((size_t)b*NCHUNK + cw)*128 + hq] - Mh[hq]);
  }
  __syncthreads();
  for (int i = t; i < NTOKS*NDIM; i += 256) { // merged attention output
    int q = i / NDIM, c = i % NDIM, h = c / HDIM;
    float s = 0.f;
    for (int cw = 0; cw < NCHUNK; ++cw)
      s += pa[((size_t)b*NCHUNK + cw)*NTOKS*NDIM + i] * big[(cw << 7) + h*16 + q];
    abuf[i] = s / Lh[h*16 + q];
  }
  __syncthreads();
  for (int i = t; i < NTOKS*NDIM; i += 256) { // x + a@Wo^T + bo
    int q = i / NDIM, o = i % NDIM;
    float s = bo[o];
    const float* ar = abuf + q*NDIM;
    const float* wr = Wo + (size_t)o*NDIM;
    for (int k = 0; k < NDIM; ++k) s += ar[k]*wr[k];
    tbuf[i] = xbuf[i] + s;
  }
  __syncthreads();
  lnorm_block(tbuf, hbuf, ln1g, ln1b, st, t);
  __syncthreads();
  for (int i = t; i < NTOKS*NFFN; i += 256) { // gelu(W1 h + b1)
    int q = i >> 10, o = i & 1023;
    float s = b1[o];
    const float* hr = hbuf + q*NDIM;
    const float* wr = W1 + (size_t)o*NDIM;
    for (int k = 0; k < NDIM; ++k) s += hr[k]*wr[k];
    big[i] = 0.5f*s*(1.f + erff(s*0.70710678118654752f));
  }
  __syncthreads();
  for (int i = t; i < NTOKS*NDIM; i += 256) { // h + W2 g + b2
    int q = i / NDIM, o = i % NDIM;
    float s = b2[o];
    const float* gr = big + (q << 10);
    const float* wr = W2 + (size_t)o*NFFN;
    for (int k = 0; k < NFFN; ++k) s += gr[k]*wr[k];
    tbuf[i] = hbuf[i] + s;
  }
  __syncthreads();
  lnorm_block(tbuf, x + (size_t)b*NTOKS*NDIM, ln2g, ln2b, st, t);
}

__global__ __launch_bounds__(256) void copy_out_kernel(
    const float* __restrict__ x, float* __restrict__ out) {
  int i = blockIdx.x*256 + threadIdx.x;
  if (i < NB*NTOKS*NDIM) out[i] = x[i];
}

// ---------------------------------------------------------------------------
extern "C" void kernel_launch(void* const* d_in, const int* in_sizes, int n_in,
                              void* d_out, int out_size, void* d_ws, size_t ws_size,
                              hipStream_t stream) {
  (void)in_sizes; (void)n_in; (void)out_size; (void)ws_size;
  const float*         kv_coords  = (const float*)d_in[0];         // [B][KV][3]
  const float*         kv_feats   = (const float*)d_in[1];         // [B][KV][240]
  const unsigned char* kv_mask    = (const unsigned char*)d_in[2]; // [B][KV] bool
  const float*         head_trans = (const float*)d_in[3];         // [B][3]
  // params flattened jax-tree style: layers (sorted leaf keys) then query_tokens
  // leaf order: 0:W1 1:W2 2:Wk 3:Wo 4:Wq 5:Wv 6:b1 7:b2 8:bk 9:bo 10:bq 11:bv
  //             12:ln1_b 13:ln1_g 14:ln2_b 15:ln2_g
  auto LP = [&](int layer, int leaf) -> const float* {
    return (const float*)d_in[4 + layer*16 + leaf];
  };
  const float* query_tokens = (const float*)d_in[4 + NLAY*16];

  // workspace carve (all offsets multiple of 256B)
  char* wsb = (char*)d_ws;
  size_t o = 0;
  float*  xs  = (float*)(wsb + o);  o += (size_t)NB*NTOKS*NDIM*4;        // 122880
  bf16_t* qBp = (bf16_t*)(wsb + o); o += (size_t)NB*NHEADS*32*16*2;      // 65536
  bf16_t* WKB = (bf16_t*)(wsb + o); o += (size_t)NDIM*NDIMP*2;           // 122880
  bf16_t* WVB = (bf16_t*)(wsb + o); o += (size_t)NDIM*NDIMP*2;           // 122880
  float*  pm  = (float*)(wsb + o);  o += (size_t)NB*NCHUNK*128*4;        // 524288
  float*  pl  = (float*)(wsb + o);  o += (size_t)NB*NCHUNK*128*4;        // 524288
  float*  pa  = (float*)(wsb + o);  o += (size_t)NB*NCHUNK*NTOKS*NDIM*4; // 15728640

  init_x_kernel<<<NB, 256, 0, stream>>>(query_tokens, xs);

  for (int L = 0; L < NLAY; ++L) {
    conv_w_kernel<<<NDIM, 256, 0, stream>>>(LP(L, 2), WKB);
    conv_w_kernel<<<NDIM, 256, 0, stream>>>(LP(L, 5), WVB);
    q_kernel<<<NB, 256, Q_LDS, stream>>>(xs, LP(L, 4), LP(L, 10),
                                         head_trans, qBp);
    attn_kernel<<<NB*WG_PER_B, 128, ATTN_LDS, stream>>>(
        kv_feats, kv_coords, kv_mask, WKB, WVB, LP(L, 8), LP(L, 11),
        qBp, pm, pl, pa);
    epilogue_kernel<<<NB, 256, EPI_LDS, stream>>>(
        pm, pl, pa, LP(L, 3), LP(L, 9), LP(L, 0), LP(L, 6), LP(L, 1),
        LP(L, 7), LP(L, 13), LP(L, 12), LP(L, 15), LP(L, 14), xs);
  }

  copy_out_kernel<<<(NB*NTOKS*NDIM + 255)/256, 256, 0, stream>>>(
      xs, (float*)d_out);
}
